// MultiHeadAttention_29755533427282
// MI455X (gfx1250) — matrix operand order repaired and checked by reading, not compiled, not run on hardware
//
#include <hip/hip_runtime.h>
#include <hip/hip_bf16.h>

typedef __attribute__((ext_vector_type(16))) _Float16 v16h;
typedef __attribute__((ext_vector_type(8)))  _Float16 v8h;
typedef __attribute__((ext_vector_type(4)))  _Float16 v4h;
typedef __attribute__((ext_vector_type(8)))  float    v8f;

union HV16 { v16h v; v8h h[2]; };
union U4H8 { uint4 u; _Float16 h[8]; };

#define DMODEL 1024
#define SEQ    2048
#define NBATCH 2
#define NHEAD  16
#define DHEAD  64
#define ROWS   (NBATCH * SEQ)   // 4096

// ---------------------------------------------------------------------------
// DPP16 rotate-reduce within rows of 16 lanes (wave32 = 2 independent rows,
// which matches the WMMA C-layout M-split exactly). Pure VALU, no LDS.
// row_ror:N ctrl encoding = 0x120 + N.
// ---------------------------------------------------------------------------
template <int CTRL>
__device__ __forceinline__ float dpp_rotf(float x) {
    return __builtin_bit_cast(float,
        __builtin_amdgcn_update_dpp(0, __builtin_bit_cast(int, x),
                                    CTRL, 0xF, 0xF, true));
}
__device__ __forceinline__ float rowmax16(float v) {
    v = fmaxf(v, dpp_rotf<0x128>(v));   // ror 8
    v = fmaxf(v, dpp_rotf<0x124>(v));   // ror 4
    v = fmaxf(v, dpp_rotf<0x122>(v));   // ror 2
    v = fmaxf(v, dpp_rotf<0x121>(v));   // ror 1
    return v;
}
__device__ __forceinline__ float rowsum16(float v) {
    v += dpp_rotf<0x128>(v);
    v += dpp_rotf<0x124>(v);
    v += dpp_rotf<0x122>(v);
    v += dpp_rotf<0x121>(v);
    return v;
}

// ---------------------------------------------------------------------------
// GEMM: Out[r][c] = sum_k X[r][k] * W[c][k] + bias[c]
// X: [ROWS][DMODEL] (f32 or f16), W: [DMODEL][DMODEL] f32 row-major,
// Out: [ROWS][DMODEL] (f16 or f32).
// Macro-tile 128x64; 8 waves, each wave = 16 rows x 64 cols (4 WMMA n-tiles).
// K staged in 32-wide chunks through LDS as f16 (pitch 40 halves).
// ---------------------------------------------------------------------------
template <typename AT, typename OT>
__global__ __launch_bounds__(256) void gemm_xwt_kernel(
    const AT* __restrict__ X, const float* __restrict__ W,
    const float* __restrict__ bias, OT* __restrict__ Out)
{
    __shared__ _Float16 Xt[128 * 40];
    __shared__ _Float16 Wt[64 * 40];
    const int tid  = threadIdx.x;
    const int lane = tid & 31, wid = tid >> 5;
    const int n    = lane & 15, halfw = lane >> 4;
    const int c0   = blockIdx.x * 64;
    const int r0   = blockIdx.y * 128;

    v8f acc[4] = {};

    for (int k0 = 0; k0 < DMODEL; k0 += 32) {
        // ---- stage X tile (convert f32 -> f16 if needed) ----
        if constexpr (sizeof(AT) == 4) {
            #pragma unroll
            for (int i = 0; i < 4; ++i) {
                int f = tid + i * 256;               // 1024 float4 total
                int row = f >> 3, kk = (f & 7) * 4;
                const float4 s = *(const float4*)&X[(size_t)(r0 + row) * DMODEL + k0 + kk];
                v4h d; d[0] = (_Float16)s.x; d[1] = (_Float16)s.y;
                       d[2] = (_Float16)s.z; d[3] = (_Float16)s.w;
                *(v4h*)&Xt[row * 40 + kk] = d;
            }
        } else {
            #pragma unroll
            for (int i = 0; i < 2; ++i) {
                int f = tid + i * 256;               // 512 uint4 total
                int row = f >> 2, kk = (f & 3) * 8;
                *(v8h*)&Xt[row * 40 + kk] =
                    *(const v8h*)&X[(size_t)(r0 + row) * DMODEL + k0 + kk];
            }
        }
        // ---- stage W tile (64 output cols x 32 k) ----
        #pragma unroll
        for (int i = 0; i < 2; ++i) {
            int f = tid + i * 256;                   // 512 float4 total
            int row = f >> 3, kk = (f & 7) * 4;
            const float4 s = *(const float4*)&W[(size_t)(c0 + row) * DMODEL + k0 + kk];
            v4h d; d[0] = (_Float16)s.x; d[1] = (_Float16)s.y;
                   d[2] = (_Float16)s.z; d[3] = (_Float16)s.w;
            *(v4h*)&Wt[row * 40 + kk] = d;
        }
        __syncthreads();

        // A-layout (16x32 f16): lane m = lane%16; halves 0-7 -> K=base1+0..7,
        // halves 8-15 -> K=base1+16..23, base1 = 0 (lanes 0-15) / 8 (lanes 16-31).
        const int base1 = halfw ? 8 : 0;
        HV16 A;
        const _Float16* xr = &Xt[(wid * 16 + n) * 40];
        A.h[0] = *(const v8h*)&xr[base1];
        A.h[1] = *(const v8h*)&xr[base1 + 16];

        // B-layout (32x16 f16): lane N = lane%16; halves 0-7 -> K=kbse+0..7,
        // halves 8-15 -> K=kbse+16..23, kbse = 0 (lanes 0-15) / 8 (lanes 16-31).
        const int kbse = halfw ? 8 : 0;
        #pragma unroll
        for (int t = 0; t < 4; ++t) {
            HV16 Bm;
            const _Float16* wr = &Wt[(t * 16 + n) * 40 + kbse];
            Bm.h[0] = *(const v8h*)&wr[0];
            Bm.h[1] = *(const v8h*)&wr[16];
            acc[t] = __builtin_amdgcn_wmma_f32_16x16x32_f16(
                false, A.v, false, Bm.v, (short)0, acc[t], false, false);
        }
        __syncthreads();
    }

    // C-layout: lane N = lane%16; VGPR r -> M = r + 8*(lane>=16).
    float bv[4];
    #pragma unroll
    for (int t = 0; t < 4; ++t) bv[t] = bias[c0 + t * 16 + n];
    #pragma unroll
    for (int r = 0; r < 8; ++r) {
        int rowg = r0 + wid * 16 + r + 8 * halfw;
        #pragma unroll
        for (int t = 0; t < 4; ++t)
            Out[(size_t)rowg * DMODEL + c0 + t * 16 + n] = (OT)(acc[t][r] + bv[t]);
    }
}

// ---------------------------------------------------------------------------
// Flash attention: grid (SEQ/128, NHEAD, NBATCH), 256 threads.
// Each wave: 16 q-rows x dh=64. Key blocks of 64 staged in LDS:
//   Kt[key][d]  natural layout, filled by GLOBAL_LOAD_ASYNC_TO_LDS_B128
//               (ASYNCcnt DMA path, no VGPR round-trip)
//   Vt[d][key]  transposed (PV-GEMM B operand needs contiguous key runs)
// P round-trips through per-wave LDS to convert C-layout -> A-layout.
// Softmax row reductions are DPP16 rotate-reduce (no ds_bpermute).
// ---------------------------------------------------------------------------
__global__ __launch_bounds__(256) void attn_kernel(
    const _Float16* __restrict__ QP, const _Float16* __restrict__ KP,
    const _Float16* __restrict__ VP, _Float16* __restrict__ OutA)
{
    __shared__ _Float16 Kt[64 * 72];
    __shared__ _Float16 Vt[64 * 72];
    __shared__ _Float16 Pt[8][16 * 72];

    const int tid  = threadIdx.x, lane = tid & 31, wid = tid >> 5;
    const int n    = lane & 15, halfw = lane >> 4;
    const int q0   = blockIdx.x * 128;
    const int h    = blockIdx.y;
    const int b    = blockIdx.z;
    const size_t rowbase = (size_t)b * SEQ;
    const int col0 = h * DHEAD;
    const int base1 = halfw ? 8 : 0;
    const int kbse  = halfw ? 8 : 0;

    // Q strip in A-layout, pre-scaled by 1/sqrt(dh) = 0.125 (exact in f16)
    HV16 Aq[2];
    {
        const _Float16* qrow = &QP[(rowbase + q0 + wid * 16 + n) * DMODEL + col0];
        #pragma unroll
        for (int c = 0; c < 2; ++c) {
            Aq[c].h[0] = *(const v8h*)&qrow[c * 32 + base1];
            Aq[c].h[1] = *(const v8h*)&qrow[c * 32 + base1 + 16];
            Aq[c].v = Aq[c].v * (_Float16)0.125f;
        }
    }

    v8f o[4] = {};
    float mrow[8], lrow[8];
    #pragma unroll
    for (int r = 0; r < 8; ++r) { mrow[r] = -1e30f; lrow[r] = 0.0f; }

    for (int kb = 0; kb < SEQ; kb += 64) {
        __syncthreads();
        // ---- stage K via async DMA to LDS; V transposed via VALU ----
        #pragma unroll
        for (int i = 0; i < 2; ++i) {
            int f = tid + i * 256;                   // 512 x 16B = 64x64 f16
            int key = f >> 3, dd = (f & 7) * 8;
            const size_t srow = (rowbase + kb + key) * DMODEL + col0 + dd;
            // K: 16B global -> LDS, tracked by ASYNCcnt (GVS addressing:
            // SGPR64 base + unsigned 32-bit VGPR byte offset)
            unsigned lds_off = (unsigned)(uintptr_t)&Kt[key * 72 + dd];
            unsigned g_off   = (unsigned)(srow * sizeof(_Float16));
            asm volatile("global_load_async_to_lds_b128 %0, %1, %2"
                         :: "v"(lds_off), "v"(g_off), "s"(KP) : "memory");
            // V: load 8 f16 along d, scatter transposed into Vt[d][key]
            U4H8 vv; vv.u = *(const uint4*)&VP[srow];
            #pragma unroll
            for (int j = 0; j < 8; ++j) Vt[(dd + j) * 72 + key] = vv.h[j];
        }
        asm volatile("s_wait_asynccnt 0" ::: "memory");
        __syncthreads();

        // ---- S = (Q*scale) @ K^T : 4 key n-tiles x 2 d-chunks ----
        v8f st[4];
        #pragma unroll
        for (int t = 0; t < 4; ++t) {
            v8f s = {};
            #pragma unroll
            for (int c = 0; c < 2; ++c) {
                HV16 Bm;
                const _Float16* kr = &Kt[(t * 16 + n) * 72 + c * 32 + kbse];
                Bm.h[0] = *(const v8h*)&kr[0];
                Bm.h[1] = *(const v8h*)&kr[16];
                s = __builtin_amdgcn_wmma_f32_16x16x32_f16(
                    false, Aq[c].v, false, Bm.v, (short)0, s, false, false);
            }
            st[t] = s;
        }

        // ---- online softmax (DPP16 row reductions; rows of 16 lanes match
        //      the C-layout half split: lanes 0-15 rows 0-7, 16-31 rows 8-15)
        #pragma unroll
        for (int r = 0; r < 8; ++r) {
            float v = st[0][r];
            v = fmaxf(v, st[1][r]); v = fmaxf(v, st[2][r]); v = fmaxf(v, st[3][r]);
            v = rowmax16(v);
            float mn = fmaxf(mrow[r], v);
            float alpha = __expf(mrow[r] - mn);
            mrow[r] = mn;
            float rs = 0.0f;
            #pragma unroll
            for (int t = 0; t < 4; ++t) {
                float p = __expf(st[t][r] - mn);
                st[t][r] = p;
                rs += p;
            }
            rs = rowsum16(rs);
            lrow[r] = lrow[r] * alpha + rs;
            #pragma unroll
            for (int t = 0; t < 4; ++t) o[t][r] *= alpha;
        }

        // ---- C-layout -> LDS -> A-layout for P (per-wave private buffer) ----
        _Float16* pbuf = Pt[wid];
        #pragma unroll
        for (int r = 0; r < 8; ++r) {
            int prow = r + 8 * halfw;
            #pragma unroll
            for (int t = 0; t < 4; ++t)
                pbuf[prow * 72 + t * 16 + n] = (_Float16)st[t][r];
        }
        asm volatile("s_wait_dscnt 0" ::: "memory");  // own-wave LDS RAW

        // ---- O += P @ V : 4 dh n-tiles x 2 key-chunks ----
        #pragma unroll
        for (int dt = 0; dt < 4; ++dt) {
            #pragma unroll
            for (int c = 0; c < 2; ++c) {
                HV16 Ap, Bm;
                const _Float16* pr = &pbuf[n * 72 + c * 32];
                Ap.h[0] = *(const v8h*)&pr[base1];
                Ap.h[1] = *(const v8h*)&pr[base1 + 16];
                const _Float16* vr = &Vt[(dt * 16 + n) * 72 + c * 32 + kbse];
                Bm.h[0] = *(const v8h*)&vr[0];
                Bm.h[1] = *(const v8h*)&vr[16];
                o[dt] = __builtin_amdgcn_wmma_f32_16x16x32_f16(
                    false, Ap.v, false, Bm.v, (short)0, o[dt], false, false);
            }
        }
    }

    // ---- normalize and store merged-head layout [B*L][D] as f16 ----
    #pragma unroll
    for (int r = 0; r < 8; ++r) {
        float inv = 1.0f / lrow[r];
        size_t rowg = rowbase + q0 + wid * 16 + r + 8 * halfw;
        #pragma unroll
        for (int dt = 0; dt < 4; ++dt)
            OutA[rowg * DMODEL + col0 + dt * 16 + n] = (_Float16)(o[dt][r] * inv);
    }
}

// ---------------------------------------------------------------------------
extern "C" void kernel_launch(void* const* d_in, const int* in_sizes, int n_in,
                              void* d_out, int out_size, void* d_ws, size_t ws_size,
                              hipStream_t stream) {
    const float* q   = (const float*)d_in[0];
    const float* k   = (const float*)d_in[1];
    const float* v   = (const float*)d_in[2];
    const float* w_q = (const float*)d_in[3];
    const float* b_q = (const float*)d_in[4];
    const float* w_k = (const float*)d_in[5];
    const float* b_k = (const float*)d_in[6];
    const float* w_v = (const float*)d_in[7];
    const float* b_v = (const float*)d_in[8];
    const float* w_o = (const float*)d_in[9];
    const float* b_o = (const float*)d_in[10];
    float* out = (float*)d_out;

    // workspace: 4 x 8 MB f16 buffers (qp, kp, vp, attn) = 32 MB
    _Float16* qp = (_Float16*)d_ws;
    _Float16* kp = qp + (size_t)ROWS * DMODEL;
    _Float16* vp = kp + (size_t)ROWS * DMODEL;
    _Float16* ap = vp + (size_t)ROWS * DMODEL;

    dim3 blk(256);
    dim3 ggrid(DMODEL / 64, ROWS / 128);
    gemm_xwt_kernel<float, _Float16><<<ggrid, blk, 0, stream>>>(q, w_q, b_q, qp);
    gemm_xwt_kernel<float, _Float16><<<ggrid, blk, 0, stream>>>(k, w_k, b_k, kp);
    gemm_xwt_kernel<float, _Float16><<<ggrid, blk, 0, stream>>>(v, w_v, b_v, vp);

    dim3 agrid(SEQ / 128, NHEAD, NBATCH);
    attn_kernel<<<agrid, blk, 0, stream>>>(qp, kp, vp, ap);

    gemm_xwt_kernel<_Float16, float><<<ggrid, blk, 0, stream>>>(ap, w_o, b_o, out);
}
